// GCN_40252433498737
// MI455X (gfx1250) — matrix operand order repaired and checked by reading, not compiled, not run on hardware
//
#include <hip/hip_runtime.h>

#define N_NODES 100000
#define N_EDGES 1600000
#define NUM_GRAPHS 256

typedef __attribute__((ext_vector_type(16))) __bf16 v16bf;
typedef __attribute__((ext_vector_type(8)))  float  v8f;

// ---------------- utility kernels ----------------

__global__ void zero_f32(float* __restrict__ p, long long n) {
    long long i = (long long)blockIdx.x * blockDim.x + threadIdx.x;
    if (i < n) p[i] = 0.0f;
}

__global__ void init_deg(float* __restrict__ deg) {
    int i = blockIdx.x * blockDim.x + threadIdx.x;
    if (i < N_NODES) deg[i] = 1.0f;   // self-loop
}

__global__ void deg_accum(const long long* __restrict__ ei, float* __restrict__ deg) {
    int e = blockIdx.x * blockDim.x + threadIdx.x;
    if (e < N_EDGES) atomicAdd(&deg[(int)ei[N_EDGES + e]], 1.0f);
}

__global__ void deg_norms(const float* __restrict__ deg,
                          float* __restrict__ isq, float* __restrict__ dinv) {
    int i = blockIdx.x * blockDim.x + threadIdx.x;
    if (i < N_NODES) {
        float d = deg[i];
        isq[i]  = rsqrtf(d);
        dinv[i] = 1.0f / d;
    }
}

// Convert W[128, HOUT] (row-major fp32) into WMMA B-fragment order so the GEMM
// reads each lane's 16 bf16 elements as one contiguous 32-byte vector:
//   Wfrag[((ct*4 + kt)*32 + lane)*16 + e] = W[k*HOUT + n]
//   n = ct*16 + (lane&15),  k = kt*32 + (lane>>4)*16 + e
template<int HOUT>
__global__ void cvt_w_frag(const float* __restrict__ w, __bf16* __restrict__ wf) {
    int i = blockIdx.x * blockDim.x + threadIdx.x;   // 0 .. 128*HOUT-1
    if (i >= 128 * HOUT) return;
    int e    = i & 15;
    int lane = (i >> 4) & 31;
    int kt   = (i >> 9) & 3;
    int ct   = i >> 11;
    int n = ct * 16 + (lane & 15);
    int k = kt * 32 + (lane >> 4) * 16 + e;
    wf[i] = (__bf16)w[k * HOUT + n];
}

// ---------------- WMMA GEMM: XL[N,HOUT] = X[N,128] * W[128,HOUT] ----------------
// One wave computes one 16x16 output tile; HOUT/16 waves per block cover a
// 16-row x HOUT-col stripe. K = 128 -> 4 x v_wmma_f32_16x16x32_bf16.
// B fragments come pre-swizzled (contiguous per lane); A is converted on the fly.

template<int HOUT>
__global__ __launch_bounds__(32 * (HOUT / 16))
void gemm_wmma(const float* __restrict__ X, const __bf16* __restrict__ Wfrag,
               float* __restrict__ XL) {
    const int lane = threadIdx.x & 31;
    const int wave = threadIdx.x >> 5;
    const int m    = lane & 15;       // row within tile (A), col within tile (B/C/D)
    const int kh   = lane >> 4;       // half-select per ISA fragment layout
    const int row  = blockIdx.x * 16 + m;

    v8f acc = {};
    const float*  xrow = X + (size_t)row * 128;
    const __bf16* wf   = Wfrag + (size_t)(wave * 4 * 32 + lane) * 16;

#pragma unroll
    for (int k0 = 0; k0 < 128; k0 += 32) {
        // A fragment (16x32 bf16): lanes 0-15 -> elems 0-15 = K k0..k0+15,
        // lanes 16-31 -> elems 0-15 = K k0+16..k0+31 (matches B fragment order).
        v16bf a;
        const float* xa = xrow + k0 + kh * 16;
#pragma unroll
        for (int e = 0; e < 8; ++e) {
            a[e]     = (__bf16)xa[e];
            a[8 + e] = (__bf16)xa[8 + e];
        }
        // B fragment: one contiguous 32-byte vector per lane per K-step.
        v16bf b = *(const v16bf*)(wf + (size_t)(k0 >> 5) * 32 * 16);

        acc = __builtin_amdgcn_wmma_f32_16x16x32_bf16(
            /*neg_a=*/false, a, /*neg_b=*/false, b,
            /*c_mod=*/(short)0, acc, /*reuse_a=*/false, /*reuse_b=*/false);
    }

    // C/D layout: lanes 0-15 -> M = r, lanes 16-31 -> M = 8 + r, N = lane%16.
    float* orow = XL + (size_t)(blockIdx.x * 16 + kh * 8) * HOUT + wave * 16 + m;
#pragma unroll
    for (int r = 0; r < 8; ++r) orow[(size_t)r * HOUT] = acc[r];
}

// ---------------- edge scatter: agg[dst] += isq[src]*isq[dst] * xl[src] ----------------
// One wave per edge; 32 lanes x (H/32) contiguous floats -> coalesced 8/16B loads,
// per-float global_atomic_add_f32 (node table lives in the 192MB L2).

template<int H>
__global__ void edge_scatter(const long long* __restrict__ ei,
                             const float* __restrict__ isq,
                             const float* __restrict__ xl,
                             float* __restrict__ agg) {
    const int lane = threadIdx.x & 31;
    const long long e = (long long)blockIdx.x * (blockDim.x >> 5) + (threadIdx.x >> 5);
    if (e >= N_EDGES) return;
    const int s = (int)ei[e];
    const int d = (int)ei[N_EDGES + e];
    const float nrm = isq[s] * isq[d];

    constexpr int PER = H / 32;
    const float* sp = xl  + (size_t)s * H + lane * PER;
    float*       dp = agg + (size_t)d * H + lane * PER;
    float v[PER];
#pragma unroll
    for (int j = 0; j < PER; ++j) v[j] = sp[j];
#pragma unroll
    for (int j = 0; j < PER; ++j) atomicAdd(&dp[j], nrm * v[j]);
}

// ---------------- fused residual-norm + bias + GELU (tanh approx) ----------------

__global__ void combine_gelu(const float* __restrict__ agg, const float* __restrict__ xl,
                             const float* __restrict__ dinv, const float* __restrict__ bias,
                             float* __restrict__ out, int H) {
    long long i = (long long)blockIdx.x * blockDim.x + threadIdx.x;
    long long total = (long long)N_NODES * H;
    if (i >= total) return;
    int node = (int)(i / H);
    int f    = (int)(i % H);
    float v = agg[i] + dinv[node] * xl[i] + bias[f];
    float t = tanhf(0.7978845608028654f * (v + 0.044715f * v * v * v));
    out[i] = 0.5f * v * (1.0f + t);
}

// ---------------- global mean pool ----------------

__global__ void pool_count(const long long* __restrict__ batch, float* __restrict__ cnt) {
    int i = blockIdx.x * blockDim.x + threadIdx.x;
    if (i < N_NODES) atomicAdd(&cnt[(int)batch[i]], 1.0f);
}

__global__ void pool_sum(const float* __restrict__ h, const long long* __restrict__ batch,
                         float* __restrict__ sums) {
    long long i = (long long)blockIdx.x * blockDim.x + threadIdx.x;
    if (i >= (long long)N_NODES * 64) return;
    int node = (int)(i >> 6);
    int f    = (int)(i & 63);
    atomicAdd(&sums[(int)batch[node] * 64 + f], h[i]);
}

__global__ void pool_div(float* __restrict__ out, const float* __restrict__ cnt) {
    int i = blockIdx.x * blockDim.x + threadIdx.x;
    if (i < NUM_GRAPHS * 64) out[i] /= fmaxf(cnt[i >> 6], 1.0f);
}

// ---------------- driver ----------------

extern "C" void kernel_launch(void* const* d_in, const int* in_sizes, int n_in,
                              void* d_out, int out_size, void* d_ws, size_t ws_size,
                              hipStream_t stream) {
    const float*     x     = (const float*)d_in[0];
    const long long* ei    = (const long long*)d_in[1];
    const long long* batch = (const long long*)d_in[2];
    const float*     W0    = (const float*)d_in[3];
    const float*     b0    = (const float*)d_in[4];
    const float*     W1    = (const float*)d_in[5];
    const float*     b1    = (const float*)d_in[6];
    const float*     W2    = (const float*)d_in[7];
    const float*     b2    = (const float*)d_in[8];
    float* out = (float*)d_out;

    char* ws = (char*)d_ws;
    float* deg  = (float*)ws; ws += sizeof(float) * N_NODES;
    float* isq  = (float*)ws; ws += sizeof(float) * N_NODES;
    float* dinv = (float*)ws; ws += sizeof(float) * N_NODES;
    float* bufB = (float*)ws; ws += sizeof(float) * (size_t)N_NODES * 128;
    float* bufC = (float*)ws; ws += sizeof(float) * (size_t)N_NODES * 128;
    float* bufD = (float*)ws; ws += sizeof(float) * (size_t)N_NODES * 128;
    __bf16* Wb  = (__bf16*)ws; ws += sizeof(__bf16) * 128 * 128;
    float* cnt  = (float*)ws; ws += sizeof(float) * NUM_GRAPHS;

    const int TB = 256;
    const long long nh128 = (long long)N_NODES * 128;
    const long long nh64  = (long long)N_NODES * 64;
    const int edgeBlocks  = (N_EDGES + 7) / 8;  // 8 waves (edges) per 256-thread block
    const int gemmBlocks  = N_NODES / 16;       // 100000 % 16 == 0

    // degrees + normalization
    init_deg<<<(N_NODES + TB - 1) / TB, TB, 0, stream>>>(deg);
    deg_accum<<<(N_EDGES + TB - 1) / TB, TB, 0, stream>>>(ei, deg);
    deg_norms<<<(N_NODES + TB - 1) / TB, TB, 0, stream>>>(deg, isq, dinv);

    // ---- layer 0: x -> bufB (H=128) ----
    cvt_w_frag<128><<<(128 * 128 + TB - 1) / TB, TB, 0, stream>>>(W0, Wb);
    gemm_wmma<128><<<gemmBlocks, 256, 0, stream>>>(x, Wb, bufB);
    zero_f32<<<(int)((nh128 + TB - 1) / TB), TB, 0, stream>>>(bufC, nh128);
    edge_scatter<128><<<edgeBlocks, 256, 0, stream>>>(ei, isq, bufB, bufC);
    combine_gelu<<<(int)((nh128 + TB - 1) / TB), TB, 0, stream>>>(bufC, bufB, dinv, b0, bufB, 128);

    // ---- layer 1: bufB -> bufC (H=128) ----
    cvt_w_frag<128><<<(128 * 128 + TB - 1) / TB, TB, 0, stream>>>(W1, Wb);
    gemm_wmma<128><<<gemmBlocks, 256, 0, stream>>>(bufB, Wb, bufC);
    zero_f32<<<(int)((nh128 + TB - 1) / TB), TB, 0, stream>>>(bufD, nh128);
    edge_scatter<128><<<edgeBlocks, 256, 0, stream>>>(ei, isq, bufC, bufD);
    combine_gelu<<<(int)((nh128 + TB - 1) / TB), TB, 0, stream>>>(bufD, bufC, dinv, b1, bufC, 128);

    // ---- layer 2: bufC -> bufB (H=64) ----
    cvt_w_frag<64><<<(128 * 64 + TB - 1) / TB, TB, 0, stream>>>(W2, Wb);
    gemm_wmma<64><<<gemmBlocks, 128, 0, stream>>>(bufC, Wb, bufB);
    zero_f32<<<(int)((nh64 + TB - 1) / TB), TB, 0, stream>>>(bufD, nh64);
    edge_scatter<64><<<edgeBlocks, 256, 0, stream>>>(ei, isq, bufB, bufD);
    combine_gelu<<<(int)((nh64 + TB - 1) / TB), TB, 0, stream>>>(bufD, bufB, dinv, b2, bufB, 64);

    // ---- global mean pool: bufB[N,64] -> out[256,64] ----
    zero_f32<<<(NUM_GRAPHS * 64 + TB - 1) / TB, TB, 0, stream>>>(out, NUM_GRAPHS * 64);
    zero_f32<<<(NUM_GRAPHS + TB - 1) / TB, TB, 0, stream>>>(cnt, NUM_GRAPHS);
    pool_count<<<(N_NODES + TB - 1) / TB, TB, 0, stream>>>(batch, cnt);
    pool_sum<<<(int)((nh64 + TB - 1) / TB), TB, 0, stream>>>(bufB, batch, out);
    pool_div<<<(NUM_GRAPHS * 64 + TB - 1) / TB, TB, 0, stream>>>(out, cnt);
}
